// PAFEM_59356448031241
// MI455X (gfx1250) — matrix operand order repaired and checked by reading, not compiled, not run on hardware
//
#include <hip/hip_runtime.h>
#include <cstdint>
#include <cstddef>

// ============================================================================
// PAFEM forward on gfx1250 (MI455X), bf16 WMMA (16x16x32) implicit-GEMM convs
// + DANet position attention. All matrix math via v_wmma_f32_16x16x32_bf16.
// Dual accumulators per wave break the WMMA RAW chain so fragment loads can
// overlap WMMA issue; global_prefetch_b8 hides latency on streamed operands.
// ============================================================================

typedef __attribute__((ext_vector_type(16))) __bf16 bf16x16;
typedef __attribute__((ext_vector_type(8)))  float  floatx8;

union FragAB {
  bf16x16 v;
  uint4   q[2];
  unsigned short us[16];
};

__device__ __forceinline__ float bf2f(unsigned short h) {
  union { unsigned u; float f; } x; x.u = ((unsigned)h) << 16; return x.f;
}
__device__ __forceinline__ unsigned short f2bf(float f) {
  union { float f; unsigned u; } x; x.f = f;
  unsigned r = x.u + 0x7FFFu + ((x.u >> 16) & 1u);
  return (unsigned short)(r >> 16);
}

// ---------------------------------------------------------------------------
// Layout conversion: x [B,C,H,W] f32 -> xb [B,HW,C] bf16 (NHWC, K contiguous)
// ---------------------------------------------------------------------------
__global__ void k_nchw2nhwc(const float* __restrict__ x, unsigned short* __restrict__ o,
                            int C, int HW, int total) {
  int idx = blockIdx.x * blockDim.x + threadIdx.x;
  if (idx >= total) return;
  int c = idx % C;
  int rem = idx / C;
  int i = rem % HW;
  int b = rem / HW;
  o[idx] = f2bf(x[((size_t)b * C + c) * HW + i]);
}

// Weights [O][I][T] f32 -> [T][O][I] bf16  (T = kh*kw taps)
__global__ void k_repack_w(const float* __restrict__ W, unsigned short* __restrict__ o,
                           int O, int I, int T, int total) {
  int idx = blockIdx.x * blockDim.x + threadIdx.x;
  if (idx >= total) return;
  int i = idx % I;
  int rem = idx / I;
  int oo = rem % O;
  int t = rem / O;
  o[idx] = f2bf(W[((size_t)oo * I + i) * T + t]);
}

// Fold conv-bias + BN into per-channel (scale, shift). has_bn==0 -> bias only.
__global__ void k_bnfuse(const float* __restrict__ bias, const float* __restrict__ g,
                         const float* __restrict__ beta, const float* __restrict__ mean,
                         const float* __restrict__ var, float* __restrict__ sc,
                         float* __restrict__ sh, int C, int has_bn) {
  int c = blockIdx.x * blockDim.x + threadIdx.x;
  if (c >= C) return;
  if (has_bn) {
    float inv = g[c] * rsqrtf(var[c] + 1e-5f);
    sc[c] = inv;
    sh[c] = (bias[c] - mean[c]) * inv + beta[c];
  } else {
    sc[c] = 1.0f;
    sh[c] = bias[c];
  }
}

// ---------------------------------------------------------------------------
// Generic implicit-GEMM conv (1x1 or 3x3 dilated, pad=dil) with fused
// scale/shift + PReLU epilogue. 4 waves/block, one 16x16 D-tile per wave.
// Cin is always a multiple of 64 here -> dual-accumulator K loop (step 64).
//   in  : [B][HW][Cin] bf16        wgt : [T][Co][Cin] bf16
//   out_mode 0: bf16 [B][HW][ostride] at channel offset ochoff
//   out_mode 1: f32  [B][Co][HW]  (final NCHW output)
//   out_mode 2: bf16 [B][Co][HW]  (transposed, for attention V)
// ---------------------------------------------------------------------------
__global__ __launch_bounds__(128) void k_wmma_conv(
    const unsigned short* __restrict__ in, const unsigned short* __restrict__ wgt,
    const float* __restrict__ scale, const float* __restrict__ shift,
    const float* __restrict__ aprm,
    unsigned short* __restrict__ outb, float* __restrict__ outf,
    int Hh, int Ww, int Cin, int Co,
    int ks, int dil, int out_mode, int ostride, int ochoff) {
  const int lane = threadIdx.x & 31;
  const int wv   = threadIdx.x >> 5;
  const int tileM = (blockIdx.x * 4 + wv) * 16;
  const int tileN = blockIdx.y * 16;
  const int HW = Hh * Ww;
  const int l15 = lane & 15;
  const int hi  = (lane >= 16) ? 1 : 0;
  const int arow = tileM + l15;
  const int ab = arow / HW;
  const int ar = arow - ab * HW;
  const int ah = ar / Ww;
  const int aw = ar - ah * Ww;
  const int kbA = hi ? 8 : 0;    // A sub-offset (ISA 16-bit A layout)
  const int kbB = hi ? 8 : 0;    // B sub-offset (same ISA layout as A)
  const int ncol = tileN + l15;
  const int pad = (ks == 3) ? dil : 0;

  floatx8 acc0 = {0.f,0.f,0.f,0.f,0.f,0.f,0.f,0.f};
  floatx8 acc1 = {0.f,0.f,0.f,0.f,0.f,0.f,0.f,0.f};

  for (int t = 0; t < ks * ks; ++t) {
    const int dh = (t / ks) * dil - pad;
    const int dw = (t % ks) * dil - pad;
    const int hh = ah + dh, wp = aw + dw;
    const bool valid = (hh >= 0) & (hh < Hh) & (wp >= 0) & (wp < Ww);
    const unsigned short* arowp = in + (size_t)((ab * Hh + hh) * Ww + wp) * (size_t)Cin;
    const unsigned short* wcolp = wgt + ((size_t)t * Co + ncol) * (size_t)Cin;
    for (int k0 = 0; k0 < Cin; k0 += 64) {
      // prefetch next K chunk of the streamed activation row
      if (valid) __builtin_prefetch(arowp + k0 + 64, 0, 1);
      FragAB A0, B0, A1, B1;
      if (valid) {
        A0.q[0] = *(const uint4*)(arowp + k0 + kbA);
        A0.q[1] = *(const uint4*)(arowp + k0 + 16 + kbA);
        A1.q[0] = *(const uint4*)(arowp + k0 + 32 + kbA);
        A1.q[1] = *(const uint4*)(arowp + k0 + 48 + kbA);
      } else {
        A0.q[0] = make_uint4(0u,0u,0u,0u); A0.q[1] = make_uint4(0u,0u,0u,0u);
        A1.q[0] = make_uint4(0u,0u,0u,0u); A1.q[1] = make_uint4(0u,0u,0u,0u);
      }
      B0.q[0] = *(const uint4*)(wcolp + k0 + kbB);
      B0.q[1] = *(const uint4*)(wcolp + k0 + kbB + 16);
      B1.q[0] = *(const uint4*)(wcolp + k0 + 32 + kbB);
      B1.q[1] = *(const uint4*)(wcolp + k0 + 32 + kbB + 16);
      acc0 = __builtin_amdgcn_wmma_f32_16x16x32_bf16(false, A0.v, false, B0.v,
                                                     (short)0, acc0, false, false);
      acc1 = __builtin_amdgcn_wmma_f32_16x16x32_bf16(false, A1.v, false, B1.v,
                                                     (short)0, acc1, false, false);
    }
  }
  floatx8 acc = acc0 + acc1;

  const float sc = scale[ncol];
  const float sh = shift[ncol];
  const float al = aprm ? aprm[0] : 1.0f;
  for (int r = 0; r < 8; ++r) {
    const int m = tileM + r + hi * 8;      // D layout: VGPR r -> row r + 8*hi
    float vv = acc[r] * sc + sh;
    vv = (vv >= 0.f) ? vv : al * vv;       // PReLU (al==1 -> identity)
    const int b = m / HW;
    const int rr = m - b * HW;
    if (out_mode == 0) {
      outb[((size_t)b * HW + rr) * (size_t)ostride + ochoff + ncol] = f2bf(vv);
    } else if (out_mode == 1) {
      outf[((size_t)b * Co + ncol) * (size_t)HW + rr] = vv;
    } else {
      outb[((size_t)b * Co + ncol) * (size_t)HW + rr] = f2bf(vv);
    }
  }
}

// ---------------------------------------------------------------------------
// Attention energy: P[i][j] = sum_c q[i][c]*k[j][c], c padded 8->32 w/ zeros.
// q,k: per-batch [N][32] bf16. One WMMA per 16x16 tile (single K step).
// ---------------------------------------------------------------------------
__global__ __launch_bounds__(128) void k_wmma_energy(
    const unsigned short* __restrict__ q, const unsigned short* __restrict__ k,
    float* __restrict__ P, int N) {
  const int lane = threadIdx.x & 31;
  const int wv = threadIdx.x >> 5;
  const int tileM = (blockIdx.x * 4 + wv) * 16;
  const int tileN = blockIdx.y * 16;
  const int l15 = lane & 15;
  const int hi = (lane >= 16) ? 1 : 0;
  const int kbA = hi ? 8 : 0;
  const int kbB = hi ? 8 : 0;
  const unsigned short* ap = q + (size_t)(tileM + l15) * 32;
  const unsigned short* bp = k + (size_t)(tileN + l15) * 32;
  FragAB A, B;
  A.q[0] = *(const uint4*)(ap + kbA);
  A.q[1] = *(const uint4*)(ap + 16 + kbA);
  B.q[0] = *(const uint4*)(bp + kbB);
  B.q[1] = *(const uint4*)(bp + kbB + 16);
  floatx8 acc = {0.f,0.f,0.f,0.f,0.f,0.f,0.f,0.f};
  acc = __builtin_amdgcn_wmma_f32_16x16x32_bf16(false, A.v, false, B.v,
                                                (short)0, acc, false, false);
  for (int r = 0; r < 8; ++r)
    P[(size_t)(tileM + r + hi * 8) * N + (tileN + l15)] = acc[r];
}

// Row softmax: P f32 [N][N] -> Pbf bf16 [N][N]. One block per row.
__global__ __launch_bounds__(256) void k_softmax(const float* __restrict__ P,
                                                 unsigned short* __restrict__ Pbf, int N) {
  __shared__ float red[256];
  const int row = blockIdx.x, tid = threadIdx.x;
  const float* rp = P + (size_t)row * N;
  float mx = -3.0e38f;
  for (int j = tid; j < N; j += 256) mx = fmaxf(mx, rp[j]);
  red[tid] = mx; __syncthreads();
  for (int s = 128; s > 0; s >>= 1) { if (tid < s) red[tid] = fmaxf(red[tid], red[tid + s]); __syncthreads(); }
  mx = red[0]; __syncthreads();
  float sum = 0.f;
  for (int j = tid; j < N; j += 256) sum += __expf(rp[j] - mx);
  red[tid] = sum; __syncthreads();
  for (int s = 128; s > 0; s >>= 1) { if (tid < s) red[tid] += red[tid + s]; __syncthreads(); }
  const float inv = 1.0f / red[0];
  unsigned short* op = Pbf + (size_t)row * N;
  for (int j = tid; j < N; j += 256) op[j] = f2bf(__expf(rp[j] - mx) * inv);
}

// ---------------------------------------------------------------------------
// PV GEMM + residual: out[i][c] = gamma * sum_j P[i][j]*v[c][j] + cbuf[i][c],
// written into the concat buffer at channel offset catOff. K = N = 2304
// (multiple of 64) -> dual-accumulator loop; P rows are streamed (prefetch).
// ---------------------------------------------------------------------------
__global__ __launch_bounds__(128) void k_wmma_pv(
    const unsigned short* __restrict__ Pbf,   // [N][N] bf16
    const unsigned short* __restrict__ vmat,  // [Co][N] bf16
    const unsigned short* __restrict__ cbuf,  // [N][Co] bf16
    const float* __restrict__ gamma,
    unsigned short* __restrict__ cat,         // [N][catStride]
    int N, int Co, int catStride, int catOff) {
  const int lane = threadIdx.x & 31;
  const int wv = threadIdx.x >> 5;
  const int tileM = (blockIdx.x * 4 + wv) * 16;
  const int tileN = blockIdx.y * 16;
  const int l15 = lane & 15;
  const int hi = (lane >= 16) ? 1 : 0;
  const int kbA = hi ? 8 : 0;
  const int kbB = hi ? 8 : 0;
  const int ncol = tileN + l15;
  const unsigned short* ap = Pbf + (size_t)(tileM + l15) * N;
  const unsigned short* bp = vmat + (size_t)ncol * N;

  floatx8 acc0 = {0.f,0.f,0.f,0.f,0.f,0.f,0.f,0.f};
  floatx8 acc1 = {0.f,0.f,0.f,0.f,0.f,0.f,0.f,0.f};
  for (int k0 = 0; k0 < N; k0 += 64) {
    __builtin_prefetch(ap + k0 + 128, 0, 1);   // stream-ahead on P row
    FragAB A0, B0, A1, B1;
    A0.q[0] = *(const uint4*)(ap + k0 + kbA);
    A0.q[1] = *(const uint4*)(ap + k0 + 16 + kbA);
    A1.q[0] = *(const uint4*)(ap + k0 + 32 + kbA);
    A1.q[1] = *(const uint4*)(ap + k0 + 48 + kbA);
    B0.q[0] = *(const uint4*)(bp + k0 + kbB);
    B0.q[1] = *(const uint4*)(bp + k0 + kbB + 16);
    B1.q[0] = *(const uint4*)(bp + k0 + 32 + kbB);
    B1.q[1] = *(const uint4*)(bp + k0 + 32 + kbB + 16);
    acc0 = __builtin_amdgcn_wmma_f32_16x16x32_bf16(false, A0.v, false, B0.v,
                                                   (short)0, acc0, false, false);
    acc1 = __builtin_amdgcn_wmma_f32_16x16x32_bf16(false, A1.v, false, B1.v,
                                                   (short)0, acc1, false, false);
  }
  floatx8 acc = acc0 + acc1;
  const float g = gamma[0];
  for (int r = 0; r < 8; ++r) {
    const int m = tileM + r + hi * 8;
    float vv = g * acc[r] + bf2f(cbuf[(size_t)m * Co + ncol]);
    cat[(size_t)m * catStride + catOff + ncol] = f2bf(vv);
  }
}

// q/k 1x1 convs (K=64, N=8): tiny — plain VALU, writes [N][32] zero-padded bf16.
__global__ void k_qk(const unsigned short* __restrict__ cbuf,
                     const float* __restrict__ qW, const float* __restrict__ qb,
                     const float* __restrict__ kW, const float* __restrict__ kb,
                     unsigned short* __restrict__ qbuf, unsigned short* __restrict__ kbuf,
                     int total) {
  int idx = blockIdx.x * blockDim.x + threadIdx.x;
  if (idx >= total) return;
  const unsigned short* cp = cbuf + (size_t)idx * 64;
  float cv[64];
  #pragma unroll
  for (int c = 0; c < 64; ++c) cv[c] = bf2f(cp[c]);
  unsigned short* qo = qbuf + (size_t)idx * 32;
  unsigned short* ko = kbuf + (size_t)idx * 32;
  for (int o = 0; o < 8; ++o) {
    float s1 = qb[o], s2 = kb[o];
    #pragma unroll
    for (int c = 0; c < 64; ++c) { s1 += cv[c] * qW[o * 64 + c]; s2 += cv[c] * kW[o * 64 + c]; }
    qo[o] = f2bf(s1); ko[o] = f2bf(s2);
  }
  for (int o = 8; o < 32; ++o) { qo[o] = 0; ko[o] = 0; }
}

// Global average pool over HW: x1 [B][HW][C] bf16 -> pooled [B][C] f32
__global__ void k_pool(const unsigned short* __restrict__ x1, float* __restrict__ pooled,
                       int C, int HW, int total) {
  int idx = blockIdx.x * blockDim.x + threadIdx.x;
  if (idx >= total) return;
  int c = idx % C, b = idx / C;
  const unsigned short* p = x1 + (size_t)b * HW * C + c;
  float s = 0.f;
  for (int i = 0; i < HW; ++i) s += bf2f(p[(size_t)i * C]);
  pooled[idx] = s / (float)HW;
}

// c5: 1x1 conv on pooled [B][Cin] + BN + PReLU -> out5 [B][Co] f32
__global__ void k_c5(const float* __restrict__ pooled, const float* __restrict__ W,
                     const float* __restrict__ sc, const float* __restrict__ sh,
                     const float* __restrict__ a, float* __restrict__ out5,
                     int Cin, int Co, int total) {
  int idx = blockIdx.x * blockDim.x + threadIdx.x;
  if (idx >= total) return;
  int o = idx % Co, b = idx / Co;
  float s = 0.f;
  for (int c = 0; c < Cin; ++c) s += pooled[b * Cin + c] * W[(size_t)o * Cin + c];
  float vv = s * sc[o] + sh[o];
  vv = (vv >= 0.f) ? vv : a[0] * vv;
  out5[idx] = vv;
}

// Broadcast out5 [B][Co] into cat channels [catOff, catOff+Co)
__global__ void k_bcast(const float* __restrict__ out5, unsigned short* __restrict__ cat,
                        int HW, int Co, int catStride, int catOff, int total) {
  int idx = blockIdx.x * blockDim.x + threadIdx.x;
  if (idx >= total) return;
  int o = idx % Co;
  int rem = idx / Co;
  int i = rem % HW;
  int b = rem / HW;
  cat[((size_t)b * HW + i) * catStride + catOff + o] = f2bf(out5[b * Co + o]);
}

// ============================================================================
// Host orchestration
// ============================================================================
struct BlkIdx { int W, b, g, beta, m, v, a, qW, qb, kW, kb, vW, vb, gamma; };

extern "C" void kernel_launch(void* const* d_in, const int* in_sizes, int n_in,
                              void* d_out, int out_size, void* d_ws, size_t ws_size,
                              hipStream_t stream) {
  (void)out_size; (void)ws_size;
  const int Bn = 8, Hh = 48, Ww = 48;
  const int HW = Hh * Ww;            // 2304
  const int M  = Bn * HW;            // 18432

  // --- input index tables: detect flattening convention from x's size ---
  BlkIdx bi[7]; int xi;
  const bool insertion = (n_in > 0 && in_sizes[0] == Bn * 512 * HW);
  if (insertion) {   // setup_inputs insertion order: x, then params{down,c1,c2,c3,c4,c5,fuse}
    xi = 0;
    auto mk = [](int base, bool attn) {
      BlkIdx r; r.W = base; r.b = base + 1; r.g = base + 2; r.beta = base + 3;
      r.m = base + 4; r.v = base + 5; r.a = base + 6;
      if (attn) { r.qW = base + 7; r.qb = base + 8; r.kW = base + 9; r.kb = base + 10;
                  r.vW = base + 11; r.vb = base + 12; r.gamma = base + 13; }
      else { r.qW = r.qb = r.kW = r.kb = r.vW = r.vb = r.gamma = -1; }
      return r;
    };
    bi[0] = mk(1, false);  bi[1] = mk(8, false);  bi[2] = mk(15, true);
    bi[3] = mk(29, true);  bi[4] = mk(43, true);  bi[5] = mk(57, false);
    bi[6] = mk(64, false);
  } else {           // jax tree_leaves (sorted keys): params{c1..fuse} then x
    xi = 70;
    auto mk = [](int base, bool attn) {
      BlkIdx r; r.W = base; r.a = base + 1; r.b = base + 2; r.beta = base + 3;
      r.g = base + 4; r.m = base + 5; r.v = base + 6;
      if (attn) { r.gamma = base + 7; r.kW = base + 8; r.kb = base + 9;
                  r.qW = base + 10; r.qb = base + 11; r.vW = base + 12; r.vb = base + 13; }
      else { r.qW = r.qb = r.kW = r.kb = r.vW = r.vb = r.gamma = -1; }
      return r;
    };
    bi[1] = mk(0, false);  bi[2] = mk(7, true);   bi[3] = mk(21, true);
    bi[4] = mk(35, true);  bi[5] = mk(49, false); bi[0] = mk(56, false);
    bi[6] = mk(63, false);
  }
  auto F = [&](int i) -> const float* {
    return (i >= 0 && i < n_in) ? (const float*)d_in[i] : nullptr;
  };

  // --- workspace bump allocator (256B aligned) ---
  char* wsb = (char*)d_ws;
  size_t off = 0;
  auto alloc = [&](size_t bytes) -> void* {
    size_t cur = (off + 255) & ~(size_t)255;
    off = cur + bytes;
    return (void*)(wsb + cur);
  };
  unsigned short* xb   = (unsigned short*)alloc((size_t)M * 512 * 2);   // x NHWC bf16
  unsigned short* x1   = (unsigned short*)alloc((size_t)M * 128 * 2);   // down out
  unsigned short* cat  = (unsigned short*)alloc((size_t)M * 320 * 2);   // concat buf
  unsigned short* cbuf = (unsigned short*)alloc((size_t)M * 64 * 2);    // branch conv out
  unsigned short* qbuf = (unsigned short*)alloc((size_t)M * 32 * 2);
  unsigned short* kbuf = (unsigned short*)alloc((size_t)M * 32 * 2);
  unsigned short* vbuf = (unsigned short*)alloc((size_t)Bn * 64 * HW * 2);
  float*          Pf   = (float*)alloc((size_t)HW * HW * 4);            // energy (per batch)
  unsigned short* Pbf  = (unsigned short*)alloc((size_t)HW * HW * 2);   // softmax bf16
  unsigned short* wbD  = (unsigned short*)alloc((size_t)9 * 128 * 512 * 2);
  unsigned short* wbBr[3]; for (int i = 0; i < 3; ++i) wbBr[i] = (unsigned short*)alloc((size_t)9 * 64 * 128 * 2);
  unsigned short* wbC1 = (unsigned short*)alloc((size_t)64 * 128 * 2);
  unsigned short* wbFu = (unsigned short*)alloc((size_t)128 * 320 * 2);
  unsigned short* wbV[3];  for (int i = 0; i < 3; ++i) wbV[i] = (unsigned short*)alloc((size_t)64 * 64 * 2);
  float* scD  = (float*)alloc(128 * 4); float* shD  = (float*)alloc(128 * 4);
  float* scC1 = (float*)alloc(64 * 4);  float* shC1 = (float*)alloc(64 * 4);
  float* scBr[3], *shBr[3], *scV[3], *shV[3];
  for (int i = 0; i < 3; ++i) { scBr[i] = (float*)alloc(64 * 4); shBr[i] = (float*)alloc(64 * 4);
                                scV[i]  = (float*)alloc(64 * 4); shV[i]  = (float*)alloc(64 * 4); }
  float* scC5 = (float*)alloc(64 * 4);  float* shC5 = (float*)alloc(64 * 4);
  float* scFu = (float*)alloc(128 * 4); float* shFu = (float*)alloc(128 * 4);
  float* pooled = (float*)alloc((size_t)Bn * 128 * 4);
  float* out5   = (float*)alloc((size_t)Bn * 64 * 4);

  auto g1 = [](int total) { return dim3((unsigned)((total + 255) / 256)); };
  const dim3 b256(256), b128(128);

  // 1) x -> NHWC bf16
  { int tot = M * 512;
    k_nchw2nhwc<<<g1(tot), b256, 0, stream>>>(F(xi), xb, 512, HW, tot); }

  // 2) down: 3x3, 512->128, pad 1
  { int tot = 9 * 128 * 512;
    k_repack_w<<<g1(tot), b256, 0, stream>>>(F(bi[0].W), wbD, 128, 512, 9, tot); }
  k_bnfuse<<<1, 128, 0, stream>>>(F(bi[0].b), F(bi[0].g), F(bi[0].beta), F(bi[0].m), F(bi[0].v), scD, shD, 128, 1);
  k_wmma_conv<<<dim3(288, 8), b128, 0, stream>>>(xb, wbD, scD, shD, F(bi[0].a),
      x1, nullptr, Hh, Ww, 512, 128, 3, 1, /*mode*/0, /*ostride*/128, /*off*/0);

  // 3) c1: 1x1 128->64 -> cat channels [0,64)
  { int tot = 64 * 128;
    k_repack_w<<<g1(tot), b256, 0, stream>>>(F(bi[1].W), wbC1, 64, 128, 1, tot); }
  k_bnfuse<<<1, 64, 0, stream>>>(F(bi[1].b), F(bi[1].g), F(bi[1].beta), F(bi[1].m), F(bi[1].v), scC1, shC1, 64, 1);
  k_wmma_conv<<<dim3(288, 4), b128, 0, stream>>>(x1, wbC1, scC1, shC1, F(bi[1].a),
      cat, nullptr, Hh, Ww, 128, 64, 1, 1, 0, 320, 0);

  // 4) c5: global pool -> 1x1 -> broadcast into cat channels [256,320)
  { int tot = Bn * 128;
    k_pool<<<g1(tot), b256, 0, stream>>>(x1, pooled, 128, HW, tot); }
  k_bnfuse<<<1, 64, 0, stream>>>(F(bi[5].b), F(bi[5].g), F(bi[5].beta), F(bi[5].m), F(bi[5].v), scC5, shC5, 64, 1);
  { int tot = Bn * 64;
    k_c5<<<g1(tot), b256, 0, stream>>>(pooled, F(bi[5].W), scC5, shC5, F(bi[5].a), out5, 128, 64, tot); }
  { int tot = M * 64;
    k_bcast<<<g1(tot), b256, 0, stream>>>(out5, cat, HW, 64, 320, 256, tot); }

  // 5) attention branches c2/c3/c4 (dil 2/4/6) -> cat channels 64/128/192
  const int dils[3] = {2, 4, 6};
  const int offs[3] = {64, 128, 192};
  for (int s = 0; s < 3; ++s) {
    const BlkIdx& bk = bi[2 + s];
    { int tot = 9 * 64 * 128;
      k_repack_w<<<g1(tot), b256, 0, stream>>>(F(bk.W), wbBr[s], 64, 128, 9, tot); }
    k_bnfuse<<<1, 64, 0, stream>>>(F(bk.b), F(bk.g), F(bk.beta), F(bk.m), F(bk.v), scBr[s], shBr[s], 64, 1);
    k_wmma_conv<<<dim3(288, 4), b128, 0, stream>>>(x1, wbBr[s], scBr[s], shBr[s], F(bk.a),
        cbuf, nullptr, Hh, Ww, 128, 64, 3, dils[s], 0, 64, 0);

    // v = 1x1 conv 64->64, bias only, stored transposed [B][64][HW]
    { int tot = 64 * 64;
      k_repack_w<<<g1(tot), b256, 0, stream>>>(F(bk.vW), wbV[s], 64, 64, 1, tot); }
    k_bnfuse<<<1, 64, 0, stream>>>(F(bk.vb), nullptr, nullptr, nullptr, nullptr, scV[s], shV[s], 64, 0);
    k_wmma_conv<<<dim3(288, 4), b128, 0, stream>>>(cbuf, wbV[s], scV[s], shV[s], nullptr,
        vbuf, nullptr, Hh, Ww, 64, 64, 1, 1, /*mode*/2, 0, 0);

    // q,k (8 channels, zero-padded to 32 for WMMA K)
    k_qk<<<g1(M), b256, 0, stream>>>(cbuf, F(bk.qW), F(bk.qb), F(bk.kW), F(bk.kb), qbuf, kbuf, M);

    for (int b = 0; b < Bn; ++b) {
      k_wmma_energy<<<dim3(36, 144), b128, 0, stream>>>(
          qbuf + (size_t)b * HW * 32, kbuf + (size_t)b * HW * 32, Pf, HW);
      k_softmax<<<dim3((unsigned)HW), b256, 0, stream>>>(Pf, Pbf, HW);
      k_wmma_pv<<<dim3(36, 4), b128, 0, stream>>>(
          Pbf, vbuf + (size_t)b * 64 * HW, cbuf + (size_t)b * HW * 64,
          F(bk.gamma), cat + (size_t)b * HW * 320, HW, 64, 320, offs[s]);
    }
  }

  // 6) fuse: 1x1 320->128 over cat -> NCHW f32 output
  { int tot = 128 * 320;
    k_repack_w<<<g1(tot), b256, 0, stream>>>(F(bi[6].W), wbFu, 128, 320, 1, tot); }
  k_bnfuse<<<1, 128, 0, stream>>>(F(bi[6].b), F(bi[6].g), F(bi[6].beta), F(bi[6].m), F(bi[6].v), scFu, shFu, 128, 1);
  k_wmma_conv<<<dim3(288, 8), b128, 0, stream>>>(cat, wbFu, scFu, shFu, F(bi[6].a),
      nullptr, (float*)d_out, Hh, Ww, 320, 128, 1, 1, /*mode*/1, 0, 0);
}
